// GCN_74990128988326
// MI455X (gfx1250) — compile-verified, matrix operand order checked by reading, not run on hardware
//
#include <hip/hip_runtime.h>
#include <hip/hip_bf16.h>
#include <math.h>

typedef __attribute__((ext_vector_type(16))) _Float16 v16h;
typedef __attribute__((ext_vector_type(8)))  float    v8f;

#define NN 100000
#define NE 3200000
#define KF 128

// ---------- degree / normalization ----------
__global__ void k_init_deg(float* deg, int n) {
    int i = blockIdx.x * blockDim.x + threadIdx.x;
    if (i < n) deg[i] = 1.0f;                      // self-loop contribution
}

__global__ void k_count_deg(const long long* __restrict__ dst, float* deg, int e) {
    int i = blockIdx.x * blockDim.x + threadIdx.x;
    if (i < e) atomicAdd(&deg[(int)dst[i]], 1.0f);
}

__global__ void k_to_dinv(float* dinv, int n) {
    int i = blockIdx.x * blockDim.x + threadIdx.x;
    if (i < n) dinv[i] = rsqrtf(dinv[i]);          // deg >= 1 always (self loops)
}

// ---------- layer 1 GEMM: h1 = x @ W1  (WMMA f16 in, f32 acc) ----------
// One wave computes one 16-row tile. B (W1 padded to 16 cols) is pre-swizzled
// into the per-lane WMMA fragment layout in LDS, built branchlessly once per block.
__global__ __launch_bounds__(256) void k_gemm1_wmma(const float* __restrict__ x,
                                                    const float* __restrict__ W1,
                                                    float* __restrict__ h1) {
    __shared__ __align__(32) _Float16 bs[4][32][16];   // [k-chunk][lane][fragment j]

    // Cooperative branchless fill: 4*32*16 = 2048 halves, 8 per thread.
    for (int idx = threadIdx.x; idx < 4 * 32 * 16; idx += 256) {
        const int c = idx >> 9;              // k-chunk 0..3
        const int l = (idx >> 4) & 31;       // lane
        const int j = idx & 15;              // half index within fragment
        const int v   = j >> 1;
        const int odd = j & 1;
        const int kOff = ((v < 4) ? (2 * v) : (16 + 2 * (v - 4))) + odd;
        const int k = c * 32 + kOff + ((l >> 4) ? 8 : 0);
        const int n = l & 15;
        const float w = W1[k * 4 + (n & 3)];           // always-valid load
        bs[c][l][j] = (_Float16)((n < 4) ? w : 0.f);   // v_cndmask, no branch
    }
    __syncthreads();

    const int lane = threadIdx.x & 31;
    const int wave = threadIdx.x >> 5;
    const int tile = blockIdx.x * 8 + wave;
    const int nTiles = NN / 16;                        // 6250 exactly
    if (tile >= nTiles) return;                        // wave-uniform: EXEC all-1s inside
    const int row0 = tile * 16;
    const int half = lane >> 4;                        // 0: lanes 0-15, 1: lanes 16-31
    const int mn   = lane & 15;                        // A: row; B/C: column

    v8f cacc = {0.f, 0.f, 0.f, 0.f, 0.f, 0.f, 0.f, 0.f};

    #pragma unroll
    for (int c = 0; c < 4; ++c) {
        // A fragment: per-lane K layout = two contiguous 8-float runs
        const float* xr = x + (size_t)(row0 + mn) * KF + c * 32 + (half ? 8 : 0);
        const float4 x0 = *(const float4*)(xr +  0);
        const float4 x1 = *(const float4*)(xr +  4);
        const float4 x2 = *(const float4*)(xr + 16);
        const float4 x3 = *(const float4*)(xr + 20);
        v16h a;
        a[0]  = (_Float16)x0.x; a[1]  = (_Float16)x0.y;
        a[2]  = (_Float16)x0.z; a[3]  = (_Float16)x0.w;
        a[4]  = (_Float16)x1.x; a[5]  = (_Float16)x1.y;
        a[6]  = (_Float16)x1.z; a[7]  = (_Float16)x1.w;
        a[8]  = (_Float16)x2.x; a[9]  = (_Float16)x2.y;
        a[10] = (_Float16)x2.z; a[11] = (_Float16)x2.w;
        a[12] = (_Float16)x3.x; a[13] = (_Float16)x3.y;
        a[14] = (_Float16)x3.z; a[15] = (_Float16)x3.w;

        const v16h b = *reinterpret_cast<const v16h*>(&bs[c][lane][0]);

        cacc = __builtin_amdgcn_wmma_f32_16x16x32_f16(false, a, false, b,
                                                      (short)0, cacc, false, false);
    }

    // C/D layout: VGPR v -> M = v + 8*half, N = mn
    if (mn < 4) {
        #pragma unroll
        for (int v = 0; v < 8; ++v) {
            const int m = v + (half ? 8 : 0);
            h1[(size_t)(row0 + m) * 4 + mn] = cacc[v];
        }
    }
}

// ---------- layer 1 aggregation (4 features) ----------
__global__ void k_self1(const float* __restrict__ dinv, const float* __restrict__ h1,
                        float* __restrict__ agg1, int n) {
    int i = blockIdx.x * blockDim.x + threadIdx.x;
    if (i < n) {
        float w = dinv[i] * dinv[i];
        float4 h = ((const float4*)h1)[i];
        float4 o; o.x = w * h.x; o.y = w * h.y; o.z = w * h.z; o.w = w * h.w;
        ((float4*)agg1)[i] = o;                    // initializes agg1
    }
}

__global__ void k_scat1(const long long* __restrict__ src, const long long* __restrict__ dst,
                        const float* __restrict__ dinv, const float* __restrict__ h1,
                        float* __restrict__ agg1, int e) {
    int i = blockIdx.x * blockDim.x + threadIdx.x;
    if (i < e) {
        int s = (int)src[i], d = (int)dst[i];
        float nrm = dinv[s] * dinv[d];
        float4 h = ((const float4*)h1)[s];
        atomicAdd(&agg1[(size_t)d * 4 + 0], nrm * h.x);
        atomicAdd(&agg1[(size_t)d * 4 + 1], nrm * h.y);
        atomicAdd(&agg1[(size_t)d * 4 + 2], nrm * h.z);
        atomicAdd(&agg1[(size_t)d * 4 + 3], nrm * h.w);
    }
}

// ---------- layer 2 linear: h2 = relu(agg1 + b1) @ W2  (4 -> 2) ----------
__global__ void k_lin2(const float* __restrict__ agg1, const float* __restrict__ b1,
                       const float* __restrict__ W2, float* __restrict__ h2, int n) {
    int i = blockIdx.x * blockDim.x + threadIdx.x;
    if (i < n) {
        float4 a = ((const float4*)agg1)[i];
        float a0 = fmaxf(a.x + b1[0], 0.f), a1 = fmaxf(a.y + b1[1], 0.f);
        float a2 = fmaxf(a.z + b1[2], 0.f), a3 = fmaxf(a.w + b1[3], 0.f);
        // W2 [4,2] row-major
        float o0 = a0 * W2[0] + a1 * W2[2] + a2 * W2[4] + a3 * W2[6];
        float o1 = a0 * W2[1] + a1 * W2[3] + a2 * W2[5] + a3 * W2[7];
        float2 o; o.x = o0; o.y = o1;
        ((float2*)h2)[i] = o;
    }
}

// ---------- layer 2 aggregation (2 features) ----------
__global__ void k_self2(const float* __restrict__ dinv, const float* __restrict__ h2,
                        float* __restrict__ agg2, int n) {
    int i = blockIdx.x * blockDim.x + threadIdx.x;
    if (i < n) {
        float w = dinv[i] * dinv[i];
        float2 h = ((const float2*)h2)[i];
        float2 o; o.x = w * h.x; o.y = w * h.y;
        ((float2*)agg2)[i] = o;
    }
}

__global__ void k_scat2(const long long* __restrict__ src, const long long* __restrict__ dst,
                        const float* __restrict__ dinv, const float* __restrict__ h2,
                        float* __restrict__ agg2, int e) {
    int i = blockIdx.x * blockDim.x + threadIdx.x;
    if (i < e) {
        int s = (int)src[i], d = (int)dst[i];
        float nrm = dinv[s] * dinv[d];
        float2 h = ((const float2*)h2)[s];
        atomicAdd(&agg2[(size_t)d * 2 + 0], nrm * h.x);
        atomicAdd(&agg2[(size_t)d * 2 + 1], nrm * h.y);
    }
}

// ---------- layer 3 linear: h3 = relu(agg2 + b2) @ W3  (2 -> 1) ----------
__global__ void k_lin3(const float* __restrict__ agg2, const float* __restrict__ b2,
                       const float* __restrict__ W3, float* __restrict__ h3, int n) {
    int i = blockIdx.x * blockDim.x + threadIdx.x;
    if (i < n) {
        float2 a = ((const float2*)agg2)[i];
        float a0 = fmaxf(a.x + b2[0], 0.f), a1 = fmaxf(a.y + b2[1], 0.f);
        h3[i] = a0 * W3[0] + a1 * W3[1];
    }
}

// ---------- layer 3 aggregation (1 feature) + sigmoid ----------
__global__ void k_self3(const float* __restrict__ dinv, const float* __restrict__ h3,
                        float* __restrict__ agg3, int n) {
    int i = blockIdx.x * blockDim.x + threadIdx.x;
    if (i < n) agg3[i] = dinv[i] * dinv[i] * h3[i];
}

__global__ void k_scat3(const long long* __restrict__ src, const long long* __restrict__ dst,
                        const float* __restrict__ dinv, const float* __restrict__ h3,
                        float* __restrict__ agg3, int e) {
    int i = blockIdx.x * blockDim.x + threadIdx.x;
    if (i < e) {
        int s = (int)src[i], d = (int)dst[i];
        atomicAdd(&agg3[d], dinv[s] * dinv[d] * h3[s]);
    }
}

__global__ void k_sigmoid(const float* __restrict__ agg3, const float* __restrict__ b3,
                          float* __restrict__ out, int n) {
    int i = blockIdx.x * blockDim.x + threadIdx.x;
    if (i < n) {
        float v = agg3[i] + b3[0];
        out[i] = 1.0f / (1.0f + expf(-v));
    }
}

extern "C" void kernel_launch(void* const* d_in, const int* in_sizes, int n_in,
                              void* d_out, int out_size, void* d_ws, size_t ws_size,
                              hipStream_t stream) {
    const float*     x   = (const float*)d_in[0];
    const long long* ei  = (const long long*)d_in[1];     // int64 [2, E]
    const float*     W1  = (const float*)d_in[2];
    const float*     b1  = (const float*)d_in[3];
    const float*     W2  = (const float*)d_in[4];
    const float*     b2  = (const float*)d_in[5];
    const float*     W3  = (const float*)d_in[6];
    const float*     b3  = (const float*)d_in[7];
    float* out = (float*)d_out;

    const long long* src = ei;
    const long long* dst = ei + NE;

    // workspace layout (floats): dinv[N] h1[4N] agg1[4N] h2[2N] agg2[2N] h3[N] agg3[N]
    float* ws   = (float*)d_ws;
    float* dinv = ws;
    float* h1   = ws + (size_t)NN;
    float* agg1 = ws + (size_t)5  * NN;
    float* h2   = ws + (size_t)9  * NN;
    float* agg2 = ws + (size_t)11 * NN;
    float* h3   = ws + (size_t)13 * NN;
    float* agg3 = ws + (size_t)14 * NN;

    const int T = 256;
    const int gN = (NN + T - 1) / T;       // node-grid
    const int gE = (NE + T - 1) / T;       // edge-grid

    // normalization (deg computed in-place in dinv buffer)
    k_init_deg <<<gN, T, 0, stream>>>(dinv, NN);
    k_count_deg<<<gE, T, 0, stream>>>(dst, dinv, NE);
    k_to_dinv  <<<gN, T, 0, stream>>>(dinv, NN);

    // layer 1: WMMA GEMM then aggregate
    const int nTiles = NN / 16;            // 6250
    k_gemm1_wmma<<<(nTiles + 7) / 8, 256, 0, stream>>>(x, W1, h1);
    k_self1<<<gN, T, 0, stream>>>(dinv, h1, agg1, NN);
    k_scat1<<<gE, T, 0, stream>>>(src, dst, dinv, h1, agg1, NE);

    // layer 2
    k_lin2 <<<gN, T, 0, stream>>>(agg1, b1, W2, h2, NN);
    k_self2<<<gN, T, 0, stream>>>(dinv, h2, agg2, NN);
    k_scat2<<<gE, T, 0, stream>>>(src, dst, dinv, h2, agg2, NE);

    // layer 3
    k_lin3 <<<gN, T, 0, stream>>>(agg2, b2, W3, h3, NN);
    k_self3<<<gN, T, 0, stream>>>(dinv, h3, agg3, NN);
    k_scat3<<<gE, T, 0, stream>>>(src, dst, dinv, h3, agg3, NE);

    k_sigmoid<<<gN, T, 0, stream>>>(agg3, b3, out, NN);
}
